// EdgePredictor_24489903521883
// MI455X (gfx1250) — matrix operand order repaired and checked by reading, not compiled, not run on hardware
//
#include <hip/hip_runtime.h>
#include <hip/hip_bf16.h>

typedef __attribute__((ext_vector_type(16))) _Float16 v16h;
typedef __attribute__((ext_vector_type(8)))  float    v8f;

#define NN 2048
#define DD 64
// P = NN*(NN-1)/2 = 2,096,128 = 2047 blocks * 1024 pairs (8 waves * 8 tiles * 16)

__device__ __forceinline__ long triuS(int i) {
  // number of pairs (a,b), a<b, with a < i  (row-major upper-triangle offset)
  return (long)i * (2 * NN - 1 - i) / 2;
}

__global__ __launch_bounds__(256)
void EdgePredictor_wmma_kernel(const float* __restrict__ emb,
                               const float* __restrict__ W1, const float* __restrict__ b1,
                               const float* __restrict__ W2, const float* __restrict__ b2,
                               const float* __restrict__ We, const float* __restrict__ be,
                               const float* __restrict__ Wt, const float* __restrict__ bt,
                               float* __restrict__ out)
{
  // Weights staged in f16, pre-swizzled into WMMA B-fragment order:
  // flat = ((chunk*4 + ntile)*32 + lane)*16 + e
  //   col n = 16*ntile + (lane&15),  K = 32*chunk + 16*(lane>>4) + e
  __shared__ __align__(32) _Float16 sW1[8192];     // 128x64
  __shared__ __align__(32) _Float16 sW2[4096];     // 64x64
  __shared__ __align__(32) _Float16 sWh[1024];     // 64x16 (cols 0..8 valid, rest 0)
  __shared__ __align__(32) _Float16 sH[8][16 * 64]; // per-wave h strip, row-major 16x64
  __shared__ __align__(16) float    sOut[8][16 * 9]; // per-wave output staging

  const int tid = threadIdx.x;

  for (int idx = tid; idx < 8192; idx += 256) {
    int e = idx & 15, l = (idx >> 4) & 31, t = (idx >> 9) & 3, c = idx >> 11;
    int n = (t << 4) + (l & 15);
    int K = (c << 5) + ((l >> 4) << 4) + e;
    sW1[idx] = (_Float16)W1[K * 64 + n];
  }
  for (int idx = tid; idx < 4096; idx += 256) {
    int e = idx & 15, l = (idx >> 4) & 31, t = (idx >> 9) & 3, c = idx >> 11;
    int n = (t << 4) + (l & 15);
    int K = (c << 5) + ((l >> 4) << 4) + e;
    sW2[idx] = (_Float16)W2[K * 64 + n];
  }
  for (int idx = tid; idx < 1024; idx += 256) {
    int e = idx & 15, l = (idx >> 4) & 31, c = idx >> 9;
    int n = l & 15;
    int K = (c << 5) + ((l >> 4) << 4) + e;
    float v = 0.f;
    if (n == 0)      v = We[K];               // existence head -> column 0
    else if (n < 9)  v = Wt[K * 8 + (n - 1)]; // type head -> columns 1..8
    sWh[idx] = (_Float16)v;
  }
  __syncthreads();

  const int wave = tid >> 5, lane = tid & 31;
  const int m = lane & 15, g = lane >> 4;
  const int n = lane & 15;

  // Loop-invariant per-lane bias values (lane owns C column n, per N-tile)
  float bv1[4], bv2[4];
  #pragma unroll
  for (int t = 0; t < 4; ++t) { bv1[t] = b1[(t << 4) + n]; bv2[t] = b2[(t << 4) + n]; }
  const float bvh = (n == 0) ? be[0] : ((n < 9) ? bt[n - 1] : 0.f);

  const long waveBase = (long)blockIdx.x * 1024 + (long)wave * 128;

  for (int mt = 0; mt < 8; ++mt) {
    const long pbase = waveBase + (long)mt * 16;
    const long p = pbase + m;          // this lane's pair (lanes l and l+16 share it)

    // Invert row-major upper-triangle enumeration: p -> (i, j), i < j
    double a = (double)NN - 0.5;
    double disc = a * a - 2.0 * (double)p;
    if (disc < 0.0) disc = 0.0;
    int i = (int)(a - sqrt(disc));
    if (i < 0) i = 0;
    if (i > NN - 2) i = NN - 2;
    while (i > 0 && triuS(i) > p) --i;
    while (i < NN - 2 && triuS(i + 1) <= p) ++i;
    const int j = (int)(p - triuS(i)) + i + 1;

    const float* xi = emb + (long)i * DD;
    const float* xj = emb + (long)j * DD;

    // A fragments for GEMM1: 16x128 (pairs x [x_i | x_j]) as 4 K-chunks of 32.
    // 16-bit A layout: lane (m, g): element e -> K = 32c + 16g + e
    v16h A1[4];
    #pragma unroll
    for (int c = 0; c < 4; ++c) {
      #pragma unroll
      for (int e = 0; e < 16; ++e) {
        int f = (c << 5) + (g << 4) + e;
        float v = (f < DD) ? xi[f] : xj[f - DD];
        A1[c][e] = (_Float16)v;
      }
    }

    // ---- GEMM1: [16x128] @ [128x64] + b1, ReLU -> sH ----
    #pragma unroll
    for (int t = 0; t < 4; ++t) {
      v8f acc = {};
      #pragma unroll
      for (int k = 0; k < 4; ++k) {
        const v16h* bp = (const v16h*)&sW1[(((k << 2) + t) << 9) + (lane << 4)];
        acc = __builtin_amdgcn_wmma_f32_16x16x32_f16(false, A1[k], false, *bp,
                                                     (short)0, acc, false, false);
      }
      int nglob = (t << 4) + n;   // C layout: lane holds column nglob, rows r + 8g
      #pragma unroll
      for (int r = 0; r < 8; ++r) {
        float v = acc[r] + bv1[t];
        v = v > 0.f ? v : 0.f;
        sH[wave][((r + (g << 3)) << 6) + nglob] = (_Float16)v;
      }
    }

    // A fragments for GEMM2 from sH (same-wave LDS ops are in-order: no barrier)
    v16h A2[2];
    #pragma unroll
    for (int c = 0; c < 2; ++c)
      #pragma unroll
      for (int e = 0; e < 16; ++e) {
        int K = (c << 5) + (g << 4) + e;
        A2[c][e] = sH[wave][(m << 6) + K];
      }

    // ---- GEMM2: [16x64] @ [64x64] + b2, ReLU -> sH (overwrite) ----
    #pragma unroll
    for (int t = 0; t < 4; ++t) {
      v8f acc = {};
      #pragma unroll
      for (int k = 0; k < 2; ++k) {
        const v16h* bp = (const v16h*)&sW2[(((k << 2) + t) << 9) + (lane << 4)];
        acc = __builtin_amdgcn_wmma_f32_16x16x32_f16(false, A2[k], false, *bp,
                                                     (short)0, acc, false, false);
      }
      int nglob = (t << 4) + n;
      #pragma unroll
      for (int r = 0; r < 8; ++r) {
        float v = acc[r] + bv2[t];
        v = v > 0.f ? v : 0.f;
        sH[wave][((r + (g << 3)) << 6) + nglob] = (_Float16)v;
      }
    }

    // ---- Head GEMM: [16x64] @ [64x16(pad)] ; cols 0..8 = [existence | type] ----
    v16h A3[2];
    #pragma unroll
    for (int c = 0; c < 2; ++c)
      #pragma unroll
      for (int e = 0; e < 16; ++e) {
        int K = (c << 5) + (g << 4) + e;
        A3[c][e] = sH[wave][(m << 6) + K];
      }

    v8f acc3 = {};
    #pragma unroll
    for (int k = 0; k < 2; ++k) {
      const v16h* bp = (const v16h*)&sWh[(k << 9) + (lane << 4)];
      acc3 = __builtin_amdgcn_wmma_f32_16x16x32_f16(false, A3[k], false, *bp,
                                                    (short)0, acc3, false, false);
    }

    // Stage 16x9 result tile in LDS, then fully coalesced global stores
    if (n < 9) {
      #pragma unroll
      for (int r = 0; r < 8; ++r)
        sOut[wave][(r + (g << 3)) * 9 + n] = acc3[r] + bvh;
    }
    const long outBase = pbase * 9;
    #pragma unroll
    for (int f = lane; f < 144; f += 32)
      out[outBase + f] = sOut[wave][f];
  }
}

extern "C" void kernel_launch(void* const* d_in, const int* in_sizes, int n_in,
                              void* d_out, int out_size, void* d_ws, size_t ws_size,
                              hipStream_t stream) {
  const float* emb = (const float*)d_in[0];
  const float* W1  = (const float*)d_in[1];
  const float* b1  = (const float*)d_in[2];
  const float* W2  = (const float*)d_in[3];
  const float* b2  = (const float*)d_in[4];
  const float* We  = (const float*)d_in[5];
  const float* be  = (const float*)d_in[6];
  const float* Wt  = (const float*)d_in[7];
  const float* bt  = (const float*)d_in[8];
  float* out = (float*)d_out;

  // P = 2048*2047/2 = 2,096,128 pairs; 1024 pairs per 256-thread block
  const int blocks = 2047;
  EdgePredictor_wmma_kernel<<<blocks, 256, 0, stream>>>(
      emb, W1, b1, W2, b2, We, be, Wt, bt, out);
}
